// Decoder_40492951666707
// MI455X (gfx1250) — matrix operand order repaired and checked by reading, not compiled, not run on hardware
//
#include <hip/hip_runtime.h>

// Problem dims (compile-time constants from the reference)
#define B_   256
#define N_   2000
#define NP_  2048           // padded node count (zero-filled rows N_..NP_-1)
#define D_   128
#define H_   8
#define DH_  16
#define CC_  131
#define SF_  3
#define TILES_ 16           // NP_/128

// ---- CDNA5 WMMA vector types ----
typedef __attribute__((ext_vector_type(16))) __bf16 v16bf;
typedef __attribute__((ext_vector_type(8)))  float  v8f;
typedef __attribute__((ext_vector_type(4)))  unsigned int u32x4;
typedef __attribute__((ext_vector_type(4)))  float  f32x4;

union V16 {                  // per-lane WMMA 16-bit operand = 32 bytes
  v16bf v;
  u32x4 q[2];
};

__device__ __forceinline__ unsigned short f2bf(float f) {
  union { float f; unsigned int i; } x; x.f = f;
  unsigned int r = x.i + 0x7fffu + ((x.i >> 16) & 1u);   // RNE
  return (unsigned short)(r >> 16);
}
__device__ __forceinline__ float bf2f(unsigned int u) {
  union { unsigned int i; float f; } x; x.i = (u & 0xffffu) << 16;
  return x.f;
}

// =====================================================================
// K1: per-batch prep.
//  - graph_embed = mean_n emb  (while streaming emb -> bf16 staging copy,
//    rows N_..NP_-1 zero-filled so the WMMA kernels need no predication)
//  - graph_ctx = ge @ W_fixed ; LayerNorm(ctx) -> W1 -> SiLU -> W2 ; query
//  - a[b,h,k] = sum_j W_node[k, 16h+j] q[16h+j] / 4  (bf16, rows 8..15 = 0)
//  - e[b,h]   = sum_j W_dyn[16h+j] q[16h+j] / 4
// =====================================================================
__global__ void k1_prep(const float* __restrict__ emb, const float* __restrict__ ctx,
                        const float* __restrict__ stf, const float* __restrict__ Wnode,
                        const float* __restrict__ Wfixed, const float* __restrict__ lng,
                        const float* __restrict__ lnb, const float* __restrict__ W1,
                        const float* __restrict__ b1, const float* __restrict__ W2,
                        const float* __restrict__ b2, const float* __restrict__ Wdctx,
                        const float* __restrict__ Wdyn,
                        unsigned short* __restrict__ embB,
                        unsigned short* __restrict__ Abf, float* __restrict__ evec)
{
  __shared__ float red[256];
  __shared__ float ge[128];
  __shared__ float lnv[CC_];
  __shared__ float hbuf[128];
  __shared__ float qbuf[128];
  __shared__ float mval, vval;

  const int b = blockIdx.x, t = threadIdx.x;
  const int d = t & 127, r = t >> 7;

  // ---- mean over N + bf16 staging (+ zero pad rows) ----
  const float* eb = emb + (size_t)b * N_ * D_;
  unsigned short* ebB = embB + (size_t)b * NP_ * D_;
  float acc = 0.f;
  for (int n = r; n < N_; n += 2) {
    float v = eb[(size_t)n * D_ + d];
    acc += v;
    ebB[(size_t)n * D_ + d] = f2bf(v);
  }
  for (int n = N_ + r; n < NP_; n += 2) ebB[(size_t)n * D_ + d] = 0;
  red[t] = acc;
  __syncthreads();
  if (t < 128) ge[t] = (red[t] + red[t + 128]) * (1.0f / (float)N_);
  __syncthreads();

  // ---- graph_ctx (kept in register for t<128) ----
  float gc = 0.f;
  if (t < 128) {
    for (int k = 0; k < 128; k++) gc += ge[k] * Wfixed[k * 128 + d];
  }

  // ---- LayerNorm over context[b, 0..130] ----
  float cv = (t < CC_) ? ctx[b * CC_ + t] : 0.f;
  red[t] = cv;
  __syncthreads();
  for (int s = 128; s > 0; s >>= 1) { if (t < s) red[t] += red[t + s]; __syncthreads(); }
  if (t == 0) mval = red[0] * (1.0f / (float)CC_);
  __syncthreads();
  red[t] = cv * cv;
  __syncthreads();
  for (int s = 128; s > 0; s >>= 1) { if (t < s) red[t] += red[t + s]; __syncthreads(); }
  if (t == 0) vval = red[0] * (1.0f / (float)CC_) - mval * mval;
  __syncthreads();
  if (t < CC_) lnv[t] = (cv - mval) * rsqrtf(vval + 1e-5f) * lng[t] + lnb[t];
  __syncthreads();

  // ---- h = silu(ln @ W1 + b1) ----
  if (t < 128) {
    float hv = b1[d];
    for (int k = 0; k < CC_; k++) hv += lnv[k] * W1[k * 128 + d];
    hbuf[d] = hv / (1.0f + __expf(-hv));     // silu
  }
  __syncthreads();

  // ---- step_ctx + query ----
  if (t < 128) {
    float st = b2[d];
    for (int k = 0; k < 128; k++) st += hbuf[k] * W2[k * 128 + d];
    float q = gc + st;
    for (int s = 0; s < SF_; s++) q += stf[b * SF_ + s] * Wdctx[s * 128 + d];
    qbuf[d] = q;
  }
  __syncthreads();

  // ---- a[b,h,k] (scaled by 1/sqrt(DH)=0.25), rows 8..15 zero ----
  for (int i = 0; i < 4; i++) {
    int id = t + 256 * i;                 // 0..1023
    int h = id >> 7, k = id & 127;
    float a = 0.f;
    for (int j = 0; j < 16; j++) a += Wnode[k * 384 + h * 16 + j] * qbuf[h * 16 + j];
    Abf[((size_t)b * 16 + h) * 128 + k] = f2bf(a * 0.25f);
    Abf[((size_t)b * 16 + 8 + h) * 128 + k] = 0;     // zero pad rows 8..15
  }
  // ---- e[b,h] ----
  if (t < H_) {
    float e = 0.f;
    for (int j = 0; j < 16; j++) e += Wdyn[t * 16 + j] * qbuf[t * 16 + j];
    evec[b * H_ + t] = e * 0.25f;
  }
}

// =====================================================================
// WMMA helper: D(16x16) += embRow_tile(16x128) @ Bcol(128x16), bf16 inputs.
// All loads unconditional (staging buffer is padded), issued up-front so
// they can be clustered; 4 chained v_wmma_f32_16x16x32_bf16.
// =====================================================================
__device__ __forceinline__ v8f wmma_dot128(const u32x4* __restrict__ arow,
                                           const u32x4* __restrict__ brow,
                                           int half)
{
  V16 A[4], Bm[4];
  #pragma unroll
  for (int c = 0; c < 4; c++) {
    A[c].q[0]  = arow[4 * c + 2 * half];      // A: row M, run 32c+16*half .. +15
    A[c].q[1]  = arow[4 * c + 2 * half + 1];
    Bm[c].q[0] = brow[4 * c + 2 * half];      // B: col N, run 32c+16*half .. +15
    Bm[c].q[1] = brow[4 * c + 2 * half + 1];
  }
  v8f acc;
  #pragma unroll
  for (int i = 0; i < 8; i++) acc[i] = 0.f;
  #pragma unroll
  for (int c = 0; c < 4; c++)
    acc = __builtin_amdgcn_wmma_f32_16x16x32_bf16(false, A[c].v, false, Bm[c].v,
                                                  (short)0, acc, false, false);
  return acc;
}

// =====================================================================
// K2: scores[b,h,n] = emb_bf16[b,n,:] . a[b,h,:] + dyn[b,n]*e[b,h]; mask -> -inf
// One wave per 16-node tile. Epilogue: the 8-node run per lane is 8-aligned
// and N_ % 8 == 0, so the bounds check is uniform -> vector loads/stores.
// =====================================================================
__global__ void k2_scores(const unsigned short* __restrict__ embB,
                          const unsigned short* __restrict__ Abf,
                          const float* __restrict__ evec,
                          const float* __restrict__ dynf,
                          const unsigned char* __restrict__ mask,
                          float* __restrict__ scores)
{
  const int b = blockIdx.y, tile = blockIdx.x, t = threadIdx.x;
  const int w = t >> 5, l = t & 31;
  const int m = l & 15, half = l >> 4;
  const int node0 = tile * 128 + w * 16;
  const int rowA = node0 + m;

  const unsigned short* ebB = embB + (size_t)b * NP_ * D_;
  const u32x4* arow = (const u32x4*)(ebB + (size_t)rowA * D_);  // 16 x u32x4 per row
  const u32x4* brow = (const u32x4*)(Abf + ((size_t)b * 16 + m) * 128);

  v8f acc = wmma_dot128(arow, brow, half);

  const int nb = node0 + half * 8;                // 8-aligned run start
  if (m < H_ && nb < N_) {                        // uniform over the run
    const float e = evec[b * H_ + m];
    const f32x4* dp = (const f32x4*)(dynf + (size_t)b * N_ + nb);
    f32x4 d0 = dp[0], d1 = dp[1];
    unsigned long long mk = *(const unsigned long long*)(mask + (size_t)b * N_ + nb);
    const float NEG = -__builtin_inff();
    f32x4 s0, s1;
    #pragma unroll
    for (int r = 0; r < 4; r++) {
      float v = acc[r] + d0[r] * e;
      s0[r] = ((mk >> (8 * r)) & 0xffull) ? NEG : v;
    }
    #pragma unroll
    for (int r = 0; r < 4; r++) {
      float v = acc[4 + r] + d1[r] * e;
      s1[r] = ((mk >> (8 * (4 + r))) & 0xffull) ? NEG : v;
    }
    f32x4* sp = (f32x4*)(scores + ((size_t)b * H_ + m) * N_ + nb);
    sp[0] = s0;
    sp[1] = s1;
  }
}

// =====================================================================
// K3: softmax over n per (b,h); writes attn bf16 in [b][n][h] layout,
// and shsum[b,h] = sum_n attn*dyn.
// =====================================================================
__global__ void k3_softmax(const float* __restrict__ scores,
                           const float* __restrict__ dynf,
                           unsigned short* __restrict__ attnB,
                           float* __restrict__ shsum)
{
  const int b = blockIdx.x >> 3, h = blockIdx.x & 7, t = threadIdx.x;
  const float* row = scores + ((size_t)b * H_ + h) * N_;
  __shared__ float red[256];
  __shared__ float smax, ssum, ssd;

  float mx = -__builtin_inff();
  for (int n = t; n < N_; n += 256) mx = fmaxf(mx, row[n]);
  red[t] = mx;
  __syncthreads();
  for (int s = 128; s > 0; s >>= 1) { if (t < s) red[t] = fmaxf(red[t], red[t + s]); __syncthreads(); }
  if (t == 0) smax = red[0];
  __syncthreads();
  mx = smax;

  float sum = 0.f, sd = 0.f;
  for (int n = t; n < N_; n += 256) {
    float p = __expf(row[n] - mx);
    sum += p;
    sd += p * dynf[(size_t)b * N_ + n];
  }
  red[t] = sum;
  __syncthreads();
  for (int s = 128; s > 0; s >>= 1) { if (t < s) red[t] += red[t + s]; __syncthreads(); }
  if (t == 0) ssum = red[0];
  __syncthreads();
  red[t] = sd;
  __syncthreads();
  for (int s = 128; s > 0; s >>= 1) { if (t < s) red[t] += red[t + s]; __syncthreads(); }
  if (t == 0) ssd = red[0];
  __syncthreads();

  float inv = 1.0f / ssum;
  for (int n = t; n < N_; n += 256)
    attnB[((size_t)b * N_ + n) * H_ + h] = f2bf(__expf(row[n] - mx) * inv);
  if (t == 0) shsum[b * H_ + h] = ssd * inv;
}

// =====================================================================
// K4: T[b,h,:] = sum_n attn[b,h,n] * emb[b,n,:]   (VALU; attn staged in LDS,
// embedding rows prefetched ahead with global_prefetch)
// =====================================================================
__global__ void k4_T(const unsigned short* __restrict__ embB,
                     const unsigned short* __restrict__ attnB,
                     float* __restrict__ Tm)
{
  __shared__ __align__(16) unsigned short sa[N_ * H_];   // 32000 B
  const int b = blockIdx.x, t = threadIdx.x;             // 128 threads, t = dim
  const unsigned short* ap = attnB + (size_t)b * N_ * H_;
  for (int i = t; i < N_ * H_; i += 128) sa[i] = ap[i];
  __syncthreads();

  const unsigned short* ebB = embB + (size_t)b * NP_ * D_;
  float a0 = 0, a1 = 0, a2 = 0, a3 = 0, a4 = 0, a5 = 0, a6 = 0, a7 = 0;
  for (int n = 0; n < N_; n++) {
    // stay ~16 rows (4 KB) ahead of the consumption point
    __builtin_prefetch(ebB + (size_t)(n + 16) * D_ + t, 0, 1);
    float e = bf2f(ebB[(size_t)n * D_ + t]);
    u32x4 av = *(const u32x4*)(sa + n * H_);             // 8 bf16 attn values
    a0 += bf2f(av[0]) * e;       a1 += bf2f(av[0] >> 16) * e;
    a2 += bf2f(av[1]) * e;       a3 += bf2f(av[1] >> 16) * e;
    a4 += bf2f(av[2]) * e;       a5 += bf2f(av[2] >> 16) * e;
    a6 += bf2f(av[3]) * e;       a7 += bf2f(av[3] >> 16) * e;
  }
  float* Tr = Tm + (size_t)b * H_ * D_;
  Tr[0 * D_ + t] = a0; Tr[1 * D_ + t] = a1; Tr[2 * D_ + t] = a2; Tr[3 * D_ + t] = a3;
  Tr[4 * D_ + t] = a4; Tr[5 * D_ + t] = a5; Tr[6 * D_ + t] = a6; Tr[7 * D_ + t] = a7;
}

// =====================================================================
// K5: heads -> glimpse -> wvec_b = (scale/sqrt(D)) * W_l @ glimpse (replicated
// across 16 B-operand columns), c_b = (scale/sqrt(D)) * W_dynL . glimpse
// =====================================================================
__global__ void k5_small(const float* __restrict__ Tm, const float* __restrict__ shsum,
                         const float* __restrict__ Wnode, const float* __restrict__ Wdyn,
                         const float* __restrict__ Wout, const float* __restrict__ scale,
                         unsigned short* __restrict__ wrep, float* __restrict__ csc)
{
  __shared__ float hv[128];
  __shared__ float gl[128];
  const int b = blockIdx.x, t = threadIdx.x;   // 128 threads
  const int h = t >> 4, j = t & 15;

  const float* Trow = Tm + ((size_t)b * H_ + h) * D_;
  float v = shsum[b * H_ + h] * Wdyn[128 + h * 16 + j];
  for (int k = 0; k < 128; k++) v += Trow[k] * Wnode[k * 384 + 128 + h * 16 + j];
  hv[t] = v;                                    // flat head vector index == t
  __syncthreads();

  float g = 0.f;
  for (int k = 0; k < 128; k++) g += hv[k] * Wout[k * 128 + t];
  gl[t] = g;
  __syncthreads();

  const float coef = scale[0] * 0.08838834764831845f;    // 1/sqrt(128)
  float wv = 0.f;
  for (int k = 0; k < 128; k++) wv += Wnode[t * 384 + 256 + k] * gl[k];
  unsigned short wb = f2bf(wv * coef);
  for (int rrow = 0; rrow < 16; rrow++)
    wrep[((size_t)b * 16 + rrow) * 128 + t] = wb;        // replicate 16 columns
  if (t == 0) {
    float c = 0.f;
    for (int k = 0; k < 128; k++) c += Wdyn[256 + k] * gl[k];
    csc[b] = c * coef;
  }
}

// =====================================================================
// K6: logits[b,n] = mask ? -1e9 : 10*tanh(emb.wvec + dyn*c)  (WMMA dot)
// =====================================================================
__global__ void k6_logits(const unsigned short* __restrict__ embB,
                          const unsigned short* __restrict__ wrep,
                          const float* __restrict__ csc,
                          const float* __restrict__ dynf,
                          const unsigned char* __restrict__ mask,
                          float* __restrict__ out)
{
  const int b = blockIdx.y, tile = blockIdx.x, t = threadIdx.x;
  const int w = t >> 5, l = t & 31;
  const int m = l & 15, half = l >> 4;
  const int node0 = tile * 128 + w * 16;
  const int rowA = node0 + m;

  const unsigned short* ebB = embB + (size_t)b * NP_ * D_;
  const u32x4* arow = (const u32x4*)(ebB + (size_t)rowA * D_);
  const u32x4* brow = (const u32x4*)(wrep + ((size_t)b * 16 + m) * 128);

  v8f acc = wmma_dot128(arow, brow, half);

  const int nb = node0 + half * 8;
  if (m == 0 && nb < N_) {            // all 16 cols identical; lanes 0 & 16 store
    const float c = csc[b];
    const f32x4* dp = (const f32x4*)(dynf + (size_t)b * N_ + nb);
    f32x4 d0 = dp[0], d1 = dp[1];
    unsigned long long mk = *(const unsigned long long*)(mask + (size_t)b * N_ + nb);
    f32x4 s0, s1;
    #pragma unroll
    for (int r = 0; r < 4; r++) {
      float lg = 10.0f * tanhf(acc[r] + d0[r] * c);
      s0[r] = ((mk >> (8 * r)) & 0xffull) ? -1e9f : lg;
    }
    #pragma unroll
    for (int r = 0; r < 4; r++) {
      float lg = 10.0f * tanhf(acc[4 + r] + d1[r] * c);
      s1[r] = ((mk >> (8 * (4 + r))) & 0xffull) ? -1e9f : lg;
    }
    f32x4* op = (f32x4*)(out + (size_t)b * N_ + nb);
    op[0] = s0;
    op[1] = s1;
  }
}

// =====================================================================
extern "C" void kernel_launch(void* const* d_in, const int* in_sizes, int n_in,
                              void* d_out, int out_size, void* d_ws, size_t ws_size,
                              hipStream_t stream)
{
  const float* emb    = (const float*)d_in[0];
  const float* ctx    = (const float*)d_in[1];
  const float* stf    = (const float*)d_in[2];
  const float* dynf   = (const float*)d_in[3];
  const unsigned char* mask = (const unsigned char*)d_in[4];   // numpy bool_ = 1 byte
  const float* Wnode  = (const float*)d_in[5];
  const float* Wfixed = (const float*)d_in[6];
  const float* lng    = (const float*)d_in[7];
  const float* lnb    = (const float*)d_in[8];
  const float* W1     = (const float*)d_in[9];
  const float* b1     = (const float*)d_in[10];
  const float* W2     = (const float*)d_in[11];
  const float* b2     = (const float*)d_in[12];
  const float* Wdctx  = (const float*)d_in[13];
  const float* Wdyn   = (const float*)d_in[14];
  const float* Wout   = (const float*)d_in[15];
  const float* scale  = (const float*)d_in[16];
  float* out = (float*)d_out;
  (void)in_sizes; (void)n_in; (void)out_size; (void)ws_size;

  char* ws = (char*)d_ws;
  size_t off = 0;
  auto take = [&](size_t bytes) -> char* {
    off = (off + 255) & ~(size_t)255;
    char* p = ws + off;
    off += bytes;
    return p;
  };
  unsigned short* embB  = (unsigned short*)take((size_t)B_ * NP_ * D_ * 2);  // 134 MB (L2-resident)
  unsigned short* Abf   = (unsigned short*)take((size_t)B_ * 16 * 128 * 2);
  float*          evec  = (float*)take((size_t)B_ * H_ * 4);
  float*          scores= (float*)take((size_t)B_ * H_ * N_ * 4);
  unsigned short* attnB = (unsigned short*)take((size_t)B_ * N_ * H_ * 2);
  float*          shsum = (float*)take((size_t)B_ * H_ * 4);
  float*          Tm    = (float*)take((size_t)B_ * H_ * D_ * 4);
  unsigned short* wrep  = (unsigned short*)take((size_t)B_ * 16 * 128 * 2);
  float*          csc   = (float*)take((size_t)B_ * 4);

  k1_prep<<<B_, 256, 0, stream>>>(emb, ctx, stf, Wnode, Wfixed, lng, lnb,
                                  W1, b1, W2, b2, Wdctx, Wdyn, embB, Abf, evec);
  k2_scores<<<dim3(TILES_, B_), 256, 0, stream>>>(embB, Abf, evec, dynf, mask, scores);
  k3_softmax<<<B_ * H_, 256, 0, stream>>>(scores, dynf, attnB, shsum);
  k4_T<<<B_, 128, 0, stream>>>(embB, attnB, Tm);
  k5_small<<<B_, 128, 0, stream>>>(Tm, shsum, Wnode, Wdyn, Wout, scale, wrep, csc);
  k6_logits<<<dim3(TILES_, B_), 256, 0, stream>>>(embB, wrep, csc, dynf, mask, out);
}
